// MOE_38792144617564
// MI455X (gfx1250) — matrix operand order repaired and checked by reading, not compiled, not run on hardware
//
#include <hip/hip_runtime.h>
#include <hip/hip_bf16.h>

// ---------------- problem constants ----------------
#define TT   8192     // B*S tokens
#define DD   1024     // d_model
#define DFFE 512      // routed expert d_ff
#define NE   16       // experts
#define FD   1024     // fused shared d_ff (2*DFF)

typedef __attribute__((ext_vector_type(16))) __bf16 v16bf;
typedef __attribute__((ext_vector_type(8)))  __bf16 v8bf;
typedef __attribute__((ext_vector_type(8)))  float  v8f;

// fp32 -> bf16 scalar (epilogue use), round-to-nearest-even
__device__ __forceinline__ __bf16 to_bf16(float f) {
  union { float f; unsigned int u; } c; c.f = f;
  unsigned int r = c.u + 0x7FFFu + ((c.u >> 16) & 1u);
  return __builtin_bit_cast(__bf16, (unsigned short)(r >> 16));
}
// packed fp32x2 -> bf16x2 (manual RNE; stored as one dword)
__device__ __forceinline__ unsigned int pk2(float a, float b) {
  union { float f; unsigned int u; } ca, cb; ca.f = a; cb.f = b;
  unsigned int ra = (ca.u + 0x7FFFu + ((ca.u >> 16) & 1u)) >> 16;
  unsigned int rb = (cb.u + 0x7FFFu + ((cb.u >> 16) & 1u)) & 0xFFFF0000u;
  return ra | rb;
}

// A-operand (16x32 bf16) per CDNA5 ISA: lane = m + 16*h
//   h=0: halves 0..7 = K 0..7,  halves 8..15 = K 16..23
//   h=1: halves 0..7 = K 8..15, halves 8..15 = K 24..31
__device__ __forceinline__ v16bf ld_a(const __bf16* base, int h) {
  v8bf lo = *(const v8bf*)(base + h * 8);
  v8bf hi = *(const v8bf*)(base + 16 + h * 8);
  return __builtin_shufflevector(lo, hi, 0,1,2,3,4,5,6,7,8,9,10,11,12,13,14,15);
}
// B-operand (32x16 bf16): lane = n + 16*h ; same K order as A (h=0 -> K 0..7,16..23; h=1 -> K 8..15,24..31)
__device__ __forceinline__ v16bf ld_b(const __bf16* base, int h) {
  v8bf lo = *(const v8bf*)(base + h * 8);
  v8bf hi = *(const v8bf*)(base + 16 + h * 8);
  return __builtin_shufflevector(lo, hi, 0,1,2,3,4,5,6,7,8,9,10,11,12,13,14,15);
}
__device__ __forceinline__ v8f wmma_bf16(v16bf a, v16bf b, v8f c) {
  return __builtin_amdgcn_wmma_f32_16x16x32_bf16(false, a, false, b, (short)0, c, false, false);
}

// ---------------- kernel 1: x -> bf16 (8 elems/thread, packed) ------------
__global__ void k_cvt_x(const float* __restrict__ x, __bf16* __restrict__ xb) {
  int i = (blockIdx.x * blockDim.x + threadIdx.x) * 8;
  float4 a = *(const float4*)(x + i);
  float4 b = *(const float4*)(x + i + 4);
  uint4 o;
  o.x = pk2(a.x, a.y); o.y = pk2(a.z, a.w);
  o.z = pk2(b.x, b.y); o.w = pk2(b.z, b.w);
  *(uint4*)(xb + i) = o;
}

// ---------------- kernel 2: gate + dispatch ----------------
// 256 thr = 8 waves, 128 tokens/block, N=16 experts (one WMMA tile per wave)
__global__ void __launch_bounds__(256)
k_gate(const __bf16* __restrict__ xb, const float* __restrict__ gate_w,
       const float* __restrict__ ebias,
       int* __restrict__ counts, int* __restrict__ list,
       int* __restrict__ te, float* __restrict__ tw) {
  extern __shared__ char smem[];
  __bf16* sGW = (__bf16*)smem;                       // 16 x 1032 halfs
  float*  sL  = (float*)(smem + 16 * 1032 * 2);      // 128 x 16
  const int tid = threadIdx.x;
  const int t0  = blockIdx.x * 128;

  for (int i = tid; i < NE * DD / 4; i += 256) {     // stage gate weights (bf16)
    int e = i >> 8, kk = (i & 255) * 4;
    float4 v = *(const float4*)(gate_w + (e << 10) + kk);
    uint2 p; p.x = pk2(v.x, v.y); p.y = pk2(v.z, v.w);
    *(uint2*)(sGW + e * 1032 + kk) = p;
  }
  __syncthreads();

  const int wv = tid >> 5, lane = tid & 31;
  const int mr = lane & 15, h = lane >> 4;
  v8f acc = {};
  const __bf16* arow = xb + (size_t)(t0 + wv * 16 + mr) * DD;
  const __bf16* brow = sGW + mr * 1032;
  for (int kc = 0; kc < DD; kc += 32)
    acc = wmma_bf16(ld_a(arow + kc, h), ld_b(brow + kc, h), acc);
#pragma unroll
  for (int r = 0; r < 8; ++r)
    sL[(wv * 16 + r + 8 * h) * 16 + mr] = acc[r];
  __syncthreads();

  if (tid < 128) {
    int t = t0 + tid;
    float lg[16];
#pragma unroll
    for (int e = 0; e < 16; ++e) lg[e] = sL[tid * 16 + e];
    int i0 = 0; float b0 = lg[0] + ebias[0];
    for (int e = 1; e < 16; ++e) { float be = lg[e] + ebias[e]; if (be > b0) { b0 = be; i0 = e; } }
    int i1 = (i0 == 0) ? 1 : 0; float bb = lg[i1] + ebias[i1];
    for (int e = 0; e < 16; ++e) {
      if (e == i0) continue;
      float be = lg[e] + ebias[e];
      if (be > bb) { bb = be; i1 = e; }
    }
    float s0 = 1.f / (1.f + __expf(-lg[i0]));
    float s1 = 1.f / (1.f + __expf(-lg[i1]));
    float inv = 1.f / (s0 + s1 + 1e-10f);
    te[t * 2 + 0] = i0; te[t * 2 + 1] = i1;
    tw[t * 2 + 0] = s0 * inv; tw[t * 2 + 1] = s1 * inv;
    int p0 = atomicAdd(&counts[i0], 1); list[i0 * TT + p0] = t * 2;
    int p1 = atomicAdd(&counts[i1], 1); list[i1 * TT + p1] = t * 2 + 1;
  }
}

// ---------------- kernel 3: routed experts (grouped GEMM, fused SwiGLU) ----
// grid (128 tiles, 16 experts); block = 64 tokens; slot buffer output.
__global__ void __launch_bounds__(256)
k_routed(const __bf16* __restrict__ xb,
         const float* __restrict__ w1, const float* __restrict__ b1,
         const float* __restrict__ w3, const float* __restrict__ b3,
         const float* __restrict__ w2,
         const int* __restrict__ counts, const int* __restrict__ list,
         float* __restrict__ slots) {
  const int e   = blockIdx.y;
  const int cnt = counts[e];
  const int m0  = blockIdx.x * 64;
  if (m0 >= cnt) return;

  extern __shared__ char smem[];
  __bf16* sH  = (__bf16*)smem;                 // 64 x 520 (H, persists)
  __bf16* sA  = (__bf16*)(smem + 66560);       // 64 x 72
  __bf16* sW1 = sA + 64 * 72;                  // 256 x 72
  __bf16* sW3 = sW1 + 256 * 72;                // 256 x 72
  __bf16* sW2 = (__bf16*)(smem + 66560);       // 512 x 40 (reuses GEMM1 staging)
  __shared__ int sTok[64];

  const int tid = threadIdx.x;
  if (tid < 64) {
    int idx = m0 + tid;
    sTok[tid] = (idx < cnt) ? list[e * TT + idx] : -1;
  }
  __syncthreads();

  const int wv = tid >> 5, lane = tid & 31;
  const int mw = wv & 3, nw = wv >> 2;
  const int mr = lane & 15, h = lane >> 4;
  const float* W1e = w1 + (size_t)e * DFFE * DD;
  const float* W3e = w3 + (size_t)e * DFFE * DD;
  const float* W2e = w2 + (size_t)e * DD * DFFE;

  // ---- GEMM1: H' = X(64xK=1024) * {w1,w3}^T, 2 N-passes of 256 cols ----
  for (int p = 0; p < 2; ++p) {
    const int n0 = p * 256;
    v8f acc1[8] = {}; v8f acc3[8] = {};
    for (int k0 = 0; k0 < DD; k0 += 64) {
      // stage A: 64x64 bf16, 16B copies
      for (int i = tid; i < 64 * 64 / 8; i += 256) {
        int r = i >> 3, kk = (i & 7) * 8;
        int t2 = sTok[r];
        v8bf v;
        if (t2 >= 0) v = *(const v8bf*)(xb + (size_t)(t2 >> 1) * DD + k0 + kk);
        else { __bf16 z = __builtin_bit_cast(__bf16, (unsigned short)0);
               v = (v8bf){z, z, z, z, z, z, z, z}; }
        *(v8bf*)(sA + r * 72 + kk) = v;
      }
      // stage W1,W3: 256x64 each, packed cvt, 8B stores
      for (int i = tid; i < 256 * 64 / 4; i += 256) {
        int r = i >> 4, kk = (i & 15) * 4;
        size_t go = (size_t)(n0 + r) * DD + k0 + kk;
        float4 v1 = *(const float4*)(W1e + go);
        float4 v3 = *(const float4*)(W3e + go);
        uint2 p1; p1.x = pk2(v1.x, v1.y); p1.y = pk2(v1.z, v1.w);
        uint2 p3; p3.x = pk2(v3.x, v3.y); p3.y = pk2(v3.z, v3.w);
        *(uint2*)(sW1 + r * 72 + kk) = p1;
        *(uint2*)(sW3 + r * 72 + kk) = p3;
      }
      __syncthreads();
#pragma unroll
      for (int ks = 0; ks < 64; ks += 32) {
        v16bf a = ld_a(sA + (mw * 16 + mr) * 72 + ks, h);
#pragma unroll
        for (int nt = 0; nt < 8; ++nt) {
          acc1[nt] = wmma_bf16(a, ld_b(sW1 + (nw * 128 + nt * 16 + mr) * 72 + ks, h), acc1[nt]);
          acc3[nt] = wmma_bf16(a, ld_b(sW3 + (nw * 128 + nt * 16 + mr) * 72 + ks, h), acc3[nt]);
        }
      }
      __syncthreads();
    }
    // SwiGLU epilogue -> sH (bf16); sH region does not overlap staging
#pragma unroll
    for (int nt = 0; nt < 8; ++nt) {
      int n = n0 + nw * 128 + nt * 16 + mr;
      float bb1 = b1[e * DFFE + n], bb3 = b3[e * DFFE + n];
#pragma unroll
      for (int r = 0; r < 8; ++r) {
        int mm = mw * 16 + r + 8 * h;
        float v1 = acc1[nt][r] + bb1, v3 = acc3[nt][r] + bb3;
        sH[mm * 520 + n] = to_bf16((v1 / (1.f + __expf(-v1))) * v3);
      }
    }
  }
  __syncthreads();

  // ---- GEMM2: Y = H(64x512) * w2^T, 2 N-passes of 512 cols ----
  for (int p = 0; p < 2; ++p) {
    const int n0 = p * 512;
    v8f acc[16] = {};
    for (int k0 = 0; k0 < DFFE; k0 += 32) {
      for (int i = tid; i < 512 * 32 / 4; i += 256) {
        int r = i >> 3, kk = (i & 7) * 4;
        float4 v = *(const float4*)(W2e + (size_t)(n0 + r) * DFFE + k0 + kk);
        uint2 pv; pv.x = pk2(v.x, v.y); pv.y = pk2(v.z, v.w);
        *(uint2*)(sW2 + r * 40 + kk) = pv;
      }
      __syncthreads();
      v16bf a = ld_a(sH + (mw * 16 + mr) * 520 + k0, h);
#pragma unroll
      for (int nt = 0; nt < 16; ++nt)
        acc[nt] = wmma_bf16(a, ld_b(sW2 + (nw * 256 + nt * 16 + mr) * 40, h), acc[nt]);
      __syncthreads();
    }
#pragma unroll
    for (int r = 0; r < 8; ++r) {
      int mm = mw * 16 + r + 8 * h;
      int t2 = sTok[mm];
      if (t2 >= 0) {
#pragma unroll
        for (int nt = 0; nt < 16; ++nt) {
          int n = n0 + nw * 256 + nt * 16 + mr;
          slots[(size_t)t2 * DD + n] = acc[nt][r];   // raw, pre-bias/pre-gate
        }
      }
    }
  }
}

// ---------------- kernel 4: shared experts + final combine ----------------
__global__ void __launch_bounds__(256)
k_shared(const __bf16* __restrict__ xb,
         const float* __restrict__ sw1, const float* __restrict__ sb1,
         const float* __restrict__ sw3, const float* __restrict__ sb3,
         const float* __restrict__ sw2, const float* __restrict__ sb2,
         const float* __restrict__ b2all, const float* __restrict__ slots,
         const int* __restrict__ te, const float* __restrict__ tw,
         float* __restrict__ out) {
  extern __shared__ char smem[];
  __bf16* sH  = (__bf16*)smem;                  // 64 x 1032
  __bf16* sA  = (__bf16*)(smem + 132096);       // 64 x 72
  __bf16* sW1 = sA + 64 * 72;                   // 256 x 72
  __bf16* sW3 = sW1 + 256 * 72;                 // 256 x 72
  __bf16* sW2 = (__bf16*)(smem + 132096);       // 512 x 40

  const int tid = threadIdx.x;
  const int t0  = blockIdx.x * 64;
  const int wv = tid >> 5, lane = tid & 31;
  const int mw = wv & 3, nw = wv >> 2;
  const int mr = lane & 15, h = lane >> 4;

  for (int p = 0; p < 4; ++p) {                 // FD=1024 over 4 N-passes
    const int n0 = p * 256;
    v8f acc1[8] = {}; v8f acc3[8] = {};
    for (int k0 = 0; k0 < DD; k0 += 64) {
      for (int i = tid; i < 64 * 64 / 8; i += 256) {
        int r = i >> 3, kk = (i & 7) * 8;
        *(v8bf*)(sA + r * 72 + kk) =
            *(const v8bf*)(xb + (size_t)(t0 + r) * DD + k0 + kk);
      }
      for (int i = tid; i < 256 * 64 / 4; i += 256) {
        int r = i >> 4, kk = (i & 15) * 4;
        size_t go = (size_t)(n0 + r) * DD + k0 + kk;
        float4 v1 = *(const float4*)(sw1 + go);
        float4 v3 = *(const float4*)(sw3 + go);
        uint2 p1; p1.x = pk2(v1.x, v1.y); p1.y = pk2(v1.z, v1.w);
        uint2 p3; p3.x = pk2(v3.x, v3.y); p3.y = pk2(v3.z, v3.w);
        *(uint2*)(sW1 + r * 72 + kk) = p1;
        *(uint2*)(sW3 + r * 72 + kk) = p3;
      }
      __syncthreads();
#pragma unroll
      for (int ks = 0; ks < 64; ks += 32) {
        v16bf a = ld_a(sA + (mw * 16 + mr) * 72 + ks, h);
#pragma unroll
        for (int nt = 0; nt < 8; ++nt) {
          acc1[nt] = wmma_bf16(a, ld_b(sW1 + (nw * 128 + nt * 16 + mr) * 72 + ks, h), acc1[nt]);
          acc3[nt] = wmma_bf16(a, ld_b(sW3 + (nw * 128 + nt * 16 + mr) * 72 + ks, h), acc3[nt]);
        }
      }
      __syncthreads();
    }
#pragma unroll
    for (int nt = 0; nt < 8; ++nt) {
      int n = n0 + nw * 128 + nt * 16 + mr;
      float bb1 = sb1[n], bb3 = sb3[n];
#pragma unroll
      for (int r = 0; r < 8; ++r) {
        int mm = mw * 16 + r + 8 * h;
        float v1 = acc1[nt][r] + bb1, v3 = acc3[nt][r] + bb3;
        sH[mm * 1032 + n] = to_bf16((v1 / (1.f + __expf(-v1))) * v3);
      }
    }
  }
  __syncthreads();

  for (int p = 0; p < 2; ++p) {                 // D=1024 out cols, 2 passes
    const int n0 = p * 512;
    v8f acc[16] = {};
    for (int k0 = 0; k0 < FD; k0 += 32) {
      for (int i = tid; i < 512 * 32 / 4; i += 256) {
        int r = i >> 3, kk = (i & 7) * 4;
        float4 v = *(const float4*)(sw2 + (size_t)(n0 + r) * FD + k0 + kk);
        uint2 pv; pv.x = pk2(v.x, v.y); pv.y = pk2(v.z, v.w);
        *(uint2*)(sW2 + r * 40 + kk) = pv;
      }
      __syncthreads();
      v16bf a = ld_a(sH + (mw * 16 + mr) * 1032 + k0, h);
#pragma unroll
      for (int nt = 0; nt < 16; ++nt)
        acc[nt] = wmma_bf16(a, ld_b(sW2 + (nw * 256 + nt * 16 + mr) * 40, h), acc[nt]);
      __syncthreads();
    }
    // fused combine: y = shared + sb2 + sum_s g_s*(slot_s + b2[e_s])
#pragma unroll
    for (int r = 0; r < 8; ++r) {
      int mm = mw * 16 + r + 8 * h;
      int t = t0 + mm;
      int e0 = te[t * 2], e1 = te[t * 2 + 1];
      float g0 = tw[t * 2], g1 = tw[t * 2 + 1];
#pragma unroll
      for (int nt = 0; nt < 16; ++nt) {
        int n = n0 + nw * 256 + nt * 16 + mr;
        float v = acc[nt][r] + sb2[n];
        v += g0 * (slots[(size_t)(t * 2 + 0) * DD + n] + b2all[e0 * DD + n]);
        v += g1 * (slots[(size_t)(t * 2 + 1) * DD + n] + b2all[e1 * DD + n]);
        out[(size_t)t * DD + n] = v;
      }
    }
  }
}

// ---------------- host launcher ----------------
extern "C" void kernel_launch(void* const* d_in, const int* in_sizes, int n_in,
                              void* d_out, int out_size, void* d_ws, size_t ws_size,
                              hipStream_t stream) {
  (void)in_sizes; (void)n_in; (void)out_size; (void)ws_size;
  const float* x      = (const float*)d_in[0];
  const float* gate_w = (const float*)d_in[1];
  const float* ebias  = (const float*)d_in[2];
  const float* w1     = (const float*)d_in[3];
  const float* b1     = (const float*)d_in[4];
  const float* w3     = (const float*)d_in[5];
  const float* b3     = (const float*)d_in[6];
  const float* w2     = (const float*)d_in[7];
  const float* b2     = (const float*)d_in[8];
  const float* sw1    = (const float*)d_in[9];
  const float* sb1    = (const float*)d_in[10];
  const float* sw3    = (const float*)d_in[11];
  const float* sb3    = (const float*)d_in[12];
  const float* sw2    = (const float*)d_in[13];
  const float* sb2    = (const float*)d_in[14];
  float* out = (float*)d_out;

  char* ws = (char*)d_ws;
  int*    counts = (int*)(ws);                               // 256 B
  int*    list   = (int*)(ws + 256);                         // E*T*4  = 512 KB
  float*  tw     = (float*)(ws + 256 + 4 * NE * TT);         // T*2*4
  int*    te     = (int*)((char*)tw + 4 * 2 * TT);           // T*2*4
  __bf16* xbf    = (__bf16*)((char*)te + 4 * 2 * TT);        // T*D*2  = 16 MB
  float*  slots  = (float*)((char*)xbf + 2 * (size_t)TT * DD); // T*2*D*4 = 64 MB

  (void)hipMemsetAsync(counts, 0, 256, stream);
  k_cvt_x<<<(TT * DD) / (8 * 256), 256, 0, stream>>>(x, xbf);
  k_gate<<<TT / 128, 256, 16 * 1032 * 2 + 128 * 16 * 4, stream>>>(
      xbf, gate_w, ebias, counts, list, te, tw);
  k_routed<<<dim3(TT / 64, NE), 256, 149504, stream>>>(
      xbf, w1, b1, w3, b3, w2, counts, list, slots);
  k_shared<<<TT / 64, 256, 215040, stream>>>(
      xbf, sw1, sb1, sw3, sb3, sw2, sb2, b2, slots, te, tw, out);
}
